// BertCrf_83391085019187
// MI455X (gfx1250) — matrix operand order repaired and checked by reading, not compiled, not run on hardware
//
#include <hip/hip_runtime.h>
#include <hip/hip_bf16.h>

#define BB   16
#define TT   512
#define HH   768
#define LL   4
#define NHH  12
#define DHH  64
#define DFFF 3072
#define NLL  9
#define BT   (BB*TT)

typedef __bf16 bf16_t;
typedef __bf16 v16bf __attribute__((ext_vector_type(16)));
typedef float  v8f   __attribute__((ext_vector_type(8)));
typedef unsigned int u32x4 __attribute__((ext_vector_type(4)));
typedef int          i32x4 __attribute__((ext_vector_type(4)));
typedef int          i32x8 __attribute__((ext_vector_type(8)));

#define GF_BT    1   // B operand given as [N,K] row-major (compute A @ B^T)
#define GF_BF16  2   // store C as bf16
#define GF_BIAS  4   // add bias[col]
#define GF_GELU  8   // tanh-approx GELU epilogue

#if __has_builtin(__builtin_amdgcn_tensor_load_to_lds) && __has_builtin(__builtin_amdgcn_s_wait_tensorcnt)
#define HAVE_TDM 1
#else
#define HAVE_TDM 0
#endif

__device__ __forceinline__ float gelu_tanh(float x) {
    const float k0 = 0.7978845608028654f, k1 = 0.044715f;
    return 0.5f * x * (1.f + tanhf(k0 * (x + k1 * x * x * x)));
}

#if HAVE_TDM
// Issue a TDM 2D tile load: tile_k elems (bf16) per row, tile_rows rows,
// global row stride row_stride elems, into LDS at lds_off. With pad=1 the TDM
// inserts 4 DWORDs of padding after every 16 DWORDs (one 32-elem bf16 row),
// matching an LDS row stride of 40 bf16.
__device__ __forceinline__ void tdm_load_2d(unsigned lds_off, const void* gaddr,
                                            int tile_k, int tile_rows,
                                            long row_stride, int pad)
{
    unsigned long long ga = (unsigned long long)gaddr;
    u32x4 g0;
    g0.x = 1u;                                            // count=1, user desc
    g0.y = lds_off;                                       // lds_addr
    g0.z = (unsigned)(ga & 0xFFFFFFFFull);                // global_addr[31:0]
    g0.w = (unsigned)((ga >> 32) & 0x1FFFFFFull) | (2u << 30);  // addr[56:32] | type=2

    i32x8 g1;
    unsigned w0 = (1u << 16);                             // data_size = 2 bytes
    if (pad) w0 |= (1u << 20) | (3u << 22) | (3u << 25);  // pad_en, 16DW interval, 4DW pad
    const unsigned td0 = (unsigned)tile_k;                // tensor_dim0 == tile
    const unsigned td1 = (unsigned)tile_rows;             // tensor_dim1 == tile
    const unsigned long long s0 = (unsigned long long)row_stride;
    g1[0] = (int)w0;
    g1[1] = (int)((td0 & 0xFFFFu) << 16);                 // dim0[15:0] @ bits63:48
    g1[2] = (int)((td0 >> 16) | ((td1 & 0xFFFFu) << 16)); // dim0[31:16], dim1[15:0]
    g1[3] = (int)((td1 >> 16) | (((unsigned)tile_k & 0xFFFFu) << 16)); // tile_dim0
    g1[4] = (int)((unsigned)tile_rows & 0xFFFFu);         // tile_dim1, tile_dim2=0
    g1[5] = (int)(unsigned)(s0 & 0xFFFFFFFFull);          // dim0_stride[31:0]
    g1[6] = (int)(unsigned)((s0 >> 32) & 0xFFFFull);      // dim0_stride[47:32]
    g1[7] = 0;
    i32x4 gz = {0, 0, 0, 0};
#if __clang_major__ >= 23
    i32x8 gz8 = {0, 0, 0, 0, 0, 0, 0, 0};
    __builtin_amdgcn_tensor_load_to_lds(g0, g1, gz, gz, gz8, 0);
#else
    __builtin_amdgcn_tensor_load_to_lds(g0, g1, gz, gz, 0);
#endif
}
#endif

// ---------------------------------------------------------------------------
// Generic tiled WMMA bf16 GEMM: C[M,N] = epilogue(A[M,K] @ B + bias)
// Block tile 128x128, K-step 32, 8 waves (wave32), each wave: 16 rows x 128 cols.
// A tile (and B tile in the BT case) staged LDS-side by the Tensor Data Mover.
// ---------------------------------------------------------------------------
__global__ __launch_bounds__(256) void gemm_bf16_wmma(
    const bf16_t* __restrict__ A, const bf16_t* __restrict__ B, void* __restrict__ C,
    const float* __restrict__ bias,
    int M, int N, int K, int lda, int ldb, int ldc,
    int zInner, long aso, long asi, long bso, long bsi, long cso, long csi,
    int flags)
{
    __shared__ __attribute__((aligned(16))) bf16_t As[128][40];
    __shared__ __attribute__((aligned(16))) bf16_t Bs[128][40];   // stored [n][k]

    const int z  = blockIdx.z;
    const int zo = z / zInner, zi = z % zInner;
    const bf16_t* Ap = A + zo * aso + zi * asi;
    const bf16_t* Bp = B + zo * bso + zi * bsi;
    const long coff  = zo * cso + zi * csi;

    const int m0  = blockIdx.y * 128;
    const int n0  = blockIdx.x * 128;
    const int tid = threadIdx.x;
    const int wave = tid >> 5, lane = tid & 31;

    v8f acc[8];
    const v8f vz = {0.f,0.f,0.f,0.f,0.f,0.f,0.f,0.f};
#pragma unroll
    for (int i = 0; i < 8; i++) acc[i] = vz;

#if HAVE_TDM
    const unsigned lds_as = (unsigned)(unsigned long long)(void*)&As[0][0];
    const unsigned lds_bs = (unsigned)(unsigned long long)(void*)&Bs[0][0];
#endif

    for (int k0 = 0; k0 < K; k0 += 32) {
        __syncthreads();   // previous iteration's fragment reads done

#if HAVE_TDM
        // ---- A tile via Tensor Data Mover (wave 0 issues the DMA) ---------
        if (wave == 0) {
            tdm_load_2d(lds_as, Ap + (long)m0 * lda + k0, 32, 128, lda, 1);
            if (flags & GF_BT)
                tdm_load_2d(lds_bs, Bp + (long)n0 * ldb + k0, 32, 128, ldb, 1);
        }
#else
        for (int c = tid; c < 512; c += 256) {
            const int r = c >> 2, cc = (c & 3) * 8;
            uint4 val = {0u,0u,0u,0u};
            const int gm = m0 + r;
            if (gm < M) val = *(const uint4*)(Ap + (long)gm * lda + (k0 + cc));
            *(uint4*)(&As[r][cc]) = val;
        }
        if (flags & GF_BT) {
            for (int c = tid; c < 512; c += 256) {
                const int n = c >> 2, cc = (c & 3) * 8;
                uint4 val = {0u,0u,0u,0u};
                if (n0 + n < N)
                    val = *(const uint4*)(Bp + (long)(n0 + n) * ldb + (k0 + cc));
                *(uint4*)(&Bs[n][cc]) = val;
            }
        }
#endif
        // ---- weight-style B ([K,N] row-major): transpose into Bs[n][k] ----
        if (!(flags & GF_BT)) {
            for (int c = tid; c < 512; c += 256) {
                const int k = c >> 4, n8 = (c & 15) * 8;
                union { uint4 u; bf16_t h[8]; } t;
                if (((ldb & 7) == 0) && (n0 + n8 + 7 < N)) {
                    t.u = *(const uint4*)(Bp + (long)(k0 + k) * ldb + (n0 + n8));
                    if (k0 + 32 < K)
                        __builtin_prefetch(Bp + (long)(k0 + 32 + k) * ldb + (n0 + n8), 0, 3);
                } else {
                    for (int j = 0; j < 8; j++)
                        t.h[j] = (n0 + n8 + j < N)
                               ? Bp[(long)(k0 + k) * ldb + (n0 + n8 + j)] : (bf16_t)0.f;
                }
                for (int j = 0; j < 8; j++) Bs[n8 + j][k] = t.h[j];
            }
        }
#if HAVE_TDM
        if (wave == 0) __builtin_amdgcn_s_wait_tensorcnt(0);
#endif
        __syncthreads();

        // ---- fragments per ISA 16-bit layouts -----------------------------
        // A 16x32: lanes 0-15 M=lane K{0..7,16..23}; lanes 16-31 M=lane-16 K{8..15,24..31}
        const int mrow = wave * 16 + (lane & 15);
        const int kb   = (lane >> 4) * 8;
        union frag_t { uint4 u[2]; v16bf v; };
        frag_t af;
        af.u[0] = *(const uint4*)(&As[mrow][kb]);
        af.u[1] = *(const uint4*)(&As[mrow][16 + kb]);
        // B 32x16: lanes 0-15 N=lane K{0..7,16..23}; lanes 16-31 N=lane-16 K{8..15,24..31}
        const int nlane = lane & 15;
        const int kb2   = (lane >> 4) * 8;
        frag_t bfm[8];
#pragma unroll
        for (int nt = 0; nt < 8; nt++) {
            bfm[nt].u[0] = *(const uint4*)(&Bs[nt * 16 + nlane][kb2]);
            bfm[nt].u[1] = *(const uint4*)(&Bs[nt * 16 + nlane][kb2 + 16]);
        }
#pragma unroll
        for (int nt = 0; nt < 8; nt++) {
            acc[nt] = __builtin_amdgcn_wmma_f32_16x16x32_bf16(
                false, af.v, false, bfm[nt].v, (short)0, acc[nt], false, false);
        }
    }

    // ---- epilogue: C row = r + 8*(lane>=16), col = lane&15 ---------------
    const int mbase = m0 + wave * 16 + ((lane >> 4) * 8);
    const int ncol  = lane & 15;
#pragma unroll
    for (int nt = 0; nt < 8; nt++) {
        const int col = n0 + nt * 16 + ncol;
        if (col >= N) continue;
        const float bv = (flags & GF_BIAS) ? bias[col] : 0.f;
#pragma unroll
        for (int r = 0; r < 8; r++) {
            const int row = mbase + r;
            if (row >= M) continue;
            float v = acc[nt][r] + bv;
            if (flags & GF_GELU) v = gelu_tanh(v);
            if (flags & GF_BF16) ((bf16_t*)C)[coff + (long)row * ldc + col] = (bf16_t)v;
            else                 ((float*)C)[coff + (long)row * ldc + col] = v;
        }
    }
}

// ---------------------------------------------------------------------------
__global__ void f32_to_bf16_kernel(const float* __restrict__ s, bf16_t* __restrict__ d, long n)
{
    long i = (long)blockIdx.x * blockDim.x + threadIdx.x;
    const long stride = (long)gridDim.x * blockDim.x;
    for (; i < n; i += stride) d[i] = (bf16_t)s[i];
}

// ---------------------------------------------------------------------------
__global__ __launch_bounds__(256) void embed_ln_kernel(
    const int* __restrict__ ids, const int* __restrict__ tti,
    const float* __restrict__ we, const float* __restrict__ pe, const float* __restrict__ te,
    const float* __restrict__ g, const float* __restrict__ b,
    float* __restrict__ xf, bf16_t* __restrict__ xb)
{
    const long tok = blockIdx.x;
    const int  t   = (int)(tok % TT);
    const int  tid = threadIdx.x;
    const int  id  = ids[tok];
    const int  ty  = tti[tok];
    float v[3]; float s = 0.f, s2 = 0.f;
#pragma unroll
    for (int j = 0; j < 3; j++) {
        const int c = tid + j * 256;
        const float val = we[(long)id * HH + c] + pe[(long)t * HH + c] + te[(long)ty * HH + c];
        v[j] = val; s += val; s2 += val * val;
    }
    __shared__ float rs[256], rs2[256];
    rs[tid] = s; rs2[tid] = s2; __syncthreads();
    for (int off = 128; off > 0; off >>= 1) {
        if (tid < off) { rs[tid] += rs[tid + off]; rs2[tid] += rs2[tid + off]; }
        __syncthreads();
    }
    const float mean = rs[0] / HH;
    const float var  = rs2[0] / HH - mean * mean;
    const float inv  = rsqrtf(var + 1e-12f);
#pragma unroll
    for (int j = 0; j < 3; j++) {
        const int c = tid + j * 256;
        const float o = (v[j] - mean) * inv * g[c] + b[c];
        xf[tok * HH + c] = o;
        xb[tok * HH + c] = (bf16_t)o;
    }
}

// x = LN(x + y) (y already has bias), also emit bf16 copy
__global__ __launch_bounds__(256) void residual_ln_kernel(
    const float* __restrict__ y, const float* __restrict__ g, const float* __restrict__ b,
    float* __restrict__ xf, bf16_t* __restrict__ xb)
{
    const long tok = blockIdx.x;
    const int  tid = threadIdx.x;
    float v[3]; float s = 0.f, s2 = 0.f;
#pragma unroll
    for (int j = 0; j < 3; j++) {
        const int c = tid + j * 256;
        const float val = xf[tok * HH + c] + y[tok * HH + c];
        v[j] = val; s += val; s2 += val * val;
    }
    __shared__ float rs[256], rs2[256];
    rs[tid] = s; rs2[tid] = s2; __syncthreads();
    for (int off = 128; off > 0; off >>= 1) {
        if (tid < off) { rs[tid] += rs[tid + off]; rs2[tid] += rs2[tid + off]; }
        __syncthreads();
    }
    const float mean = rs[0] / HH;
    const float var  = rs2[0] / HH - mean * mean;
    const float inv  = rsqrtf(var + 1e-12f);
#pragma unroll
    for (int j = 0; j < 3; j++) {
        const int c = tid + j * 256;
        const float o = (v[j] - mean) * inv * g[c] + b[c];
        xf[tok * HH + c] = o;
        xb[tok * HH + c] = (bf16_t)o;
    }
}

// softmax over keys; one block per (b, h, q) row of 512 scores
__global__ __launch_bounds__(256) void softmax_kernel(
    const float* __restrict__ S, bf16_t* __restrict__ P,
    const int* __restrict__ am, float scale)
{
    const long row  = blockIdx.x;
    const int  bidx = (int)(row / ((long)NHH * TT));
    const int  tid  = threadIdx.x;
    const float* sp = S + row * TT;
    float v0 = sp[tid]       * scale + (am[bidx * TT + tid]       ? 0.f : -1e9f);
    float v1 = sp[tid + 256] * scale + (am[bidx * TT + tid + 256] ? 0.f : -1e9f);
    __shared__ float red[256];
    red[tid] = fmaxf(v0, v1); __syncthreads();
    for (int off = 128; off > 0; off >>= 1) {
        if (tid < off) red[tid] = fmaxf(red[tid], red[tid + off]);
        __syncthreads();
    }
    const float mx = red[0]; __syncthreads();
    const float e0 = __expf(v0 - mx), e1 = __expf(v1 - mx);
    red[tid] = e0 + e1; __syncthreads();
    for (int off = 128; off > 0; off >>= 1) {
        if (tid < off) red[tid] += red[tid + off];
        __syncthreads();
    }
    const float inv = 1.f / red[0];
    P[row * TT + tid]       = (bf16_t)(e0 * inv);
    P[row * TT + tid + 256] = (bf16_t)(e1 * inv);
}

// ---------------------------------------------------------------------------
// CRF: numerator, log-partition, Viterbi. One thread per batch element.
// out[0] = loss; out[1 + b*511 + t] = predicted tag (as float).
// ---------------------------------------------------------------------------
__global__ __launch_bounds__(32) void crf_kernel(
    const float* __restrict__ logits, const int* __restrict__ labels,
    const int* __restrict__ am,
    const float* __restrict__ stt, const float* __restrict__ endt,
    const float* __restrict__ trn,
    float* __restrict__ out, signed char* __restrict__ hist)
{
    __shared__ float str[NLL * NLL], ss[NLL], se[NLL], llh[BB];
    const int tid = threadIdx.x;
    if (tid < NLL * NLL) str[tid] = trn[tid];
    if (tid < NLL) { ss[tid] = stt[tid]; se[tid] = endt[tid]; }
    __syncthreads();

    if (tid < BB) {
        const int b = tid;
        const float* em = logits + ((long)b * TT + 1) * NLL;   // em(t,n), t=0..510
        const int* tg = labels + b * TT + 1;
        const int* mk = am     + b * TT + 1;

        // numerator
        int prev = tg[0];
        float num = ss[prev] + em[prev];
        for (int t = 1; t <= 510; t++) {
            const float m = (float)mk[t];
            const int tgt = tg[t];
            num += m * (str[prev * NLL + tgt] + em[t * NLL + tgt]);
            if (m > 0.f) prev = tgt;
        }
        num += se[prev];

        // log-partition (forward logsumexp)
        float sc[NLL];
        for (int n = 0; n < NLL; n++) sc[n] = ss[n] + em[n];
        for (int t = 1; t <= 510; t++) {
            const float m = (float)mk[t];
            float nxt[NLL];
            for (int c = 0; c < NLL; c++) {
                float mxv = -3.4e38f;
                for (int p = 0; p < NLL; p++) mxv = fmaxf(mxv, sc[p] + str[p * NLL + c]);
                float sum = 0.f;
                for (int p = 0; p < NLL; p++) sum += __expf(sc[p] + str[p * NLL + c] - mxv);
                nxt[c] = mxv + __logf(sum) + em[t * NLL + c];
            }
            if (m > 0.f) for (int c = 0; c < NLL; c++) sc[c] = nxt[c];
        }
        float mxv = -3.4e38f;
        for (int n = 0; n < NLL; n++) mxv = fmaxf(mxv, sc[n] + se[n]);
        float sum = 0.f;
        for (int n = 0; n < NLL; n++) sum += __expf(sc[n] + se[n] - mxv);
        llh[b] = num - (mxv + __logf(sum));

        // Viterbi
        float vs[NLL];
        for (int n = 0; n < NLL; n++) vs[n] = ss[n] + em[n];
        for (int t = 1; t <= 510; t++) {
            const float m = (float)mk[t];
            float ns[NLL]; int bi[NLL];
            for (int c = 0; c < NLL; c++) {
                float best = -3.4e38f; int bp = 0;
                for (int p = 0; p < NLL; p++) {
                    const float v = vs[p] + str[p * NLL + c];
                    if (v > best) { best = v; bp = p; }
                }
                ns[c] = best + em[t * NLL + c]; bi[c] = bp;
            }
            for (int c = 0; c < NLL; c++) {
                hist[(long)(t - 1) * BB * NLL + b * NLL + c] = (signed char)bi[c];
                if (m > 0.f) vs[c] = ns[c];
            }
        }
        float best = -3.4e38f; int last = 0;
        for (int n = 0; n < NLL; n++) { const float v = vs[n] + se[n]; if (v > best) { best = v; last = n; } }
        out[1 + b * 511 + 510] = (float)last;
        int tag = last;
        for (int t = 509; t >= 0; t--) {
            tag = hist[(long)t * BB * NLL + b * NLL + tag];
            out[1 + b * 511 + t] = (float)tag;
        }
    }
    __syncthreads();
    if (tid == 0) {
        float s = 0.f;
        for (int b = 0; b < BB; b++) s += llh[b];
        out[0] = -s / BB;
    }
}

// ---------------------------------------------------------------------------
extern "C" void kernel_launch(void* const* d_in, const int* in_sizes, int n_in,
                              void* d_out, int out_size, void* d_ws, size_t ws_size,
                              hipStream_t stream)
{
    (void)in_sizes; (void)n_in; (void)out_size; (void)ws_size;

    const int*   input_ids = (const int*)  d_in[0];
    const int*   attn      = (const int*)  d_in[1];
    const int*   tti       = (const int*)  d_in[2];
    const int*   labels    = (const int*)  d_in[3];
    const float* we   = (const float*)d_in[4];
    const float* pe   = (const float*)d_in[5];
    const float* te   = (const float*)d_in[6];
    const float* egs  = (const float*)d_in[7];
    const float* egb  = (const float*)d_in[8];
    const float* Wqkv = (const float*)d_in[9];
    const float* bqkv = (const float*)d_in[10];
    const float* Wo   = (const float*)d_in[11];
    const float* bo   = (const float*)d_in[12];
    const float* l1s  = (const float*)d_in[13];
    const float* l1b  = (const float*)d_in[14];
    const float* Wff1 = (const float*)d_in[15];
    const float* bff1 = (const float*)d_in[16];
    const float* Wff2 = (const float*)d_in[17];
    const float* bff2 = (const float*)d_in[18];
    const float* l2s  = (const float*)d_in[19];
    const float* l2b  = (const float*)d_in[20];
    const float* Wcls = (const float*)d_in[21];
    const float* bcls = (const float*)d_in[22];
    const float* cst  = (const float*)d_in[23];
    const float* cen  = (const float*)d_in[24];
    const float* ctr  = (const float*)d_in[25];

    char* wp = (char*)d_ws;
    auto carve = [&](size_t bytes) -> char* {
        char* r = wp; wp += (bytes + 255) & ~(size_t)255; return r;
    };
    const long nWqkv = (long)LL * HH * 3 * HH;
    const long nWo   = (long)LL * HH * HH;
    const long nWff1 = (long)LL * HH * DFFF;
    const long nWff2 = (long)LL * DFFF * HH;
    const long nWcls = (long)HH * NLL;

    bf16_t* wqkv_b = (bf16_t*)carve(nWqkv * 2);
    bf16_t* wo_b   = (bf16_t*)carve(nWo   * 2);
    bf16_t* wff1_b = (bf16_t*)carve(nWff1 * 2);
    bf16_t* wff2_b = (bf16_t*)carve(nWff2 * 2);
    bf16_t* wcls_b = (bf16_t*)carve(nWcls * 2);
    float*  x_f    = (float*) carve((long)BT * HH * 4);
    bf16_t* x_b    = (bf16_t*)carve((long)BT * HH * 2);
    bf16_t* qkv_b  = (bf16_t*)carve((long)BT * 3 * HH * 2);
    float*  scores = (float*) carve((long)BB * NHH * TT * TT * 4);
    bf16_t* p_b    = (bf16_t*)carve((long)BB * NHH * TT * TT * 2);
    bf16_t* ctx_b  = (bf16_t*)carve((long)BT * HH * 2);
    float*  y_f    = (float*) carve((long)BT * HH * 4);
    bf16_t* h_b    = (bf16_t*)carve((long)BT * DFFF * 2);
    float*  logits = (float*) carve((long)BT * NLL * 4);
    signed char* hist = (signed char*)carve((long)510 * BB * NLL);

    auto cvt = [&](const float* s, bf16_t* d, long n) {
        const int blocks = (int)((n + 255) / 256);
        f32_to_bf16_kernel<<<blocks, 256, 0, stream>>>(s, d, n);
    };
    cvt(Wqkv, wqkv_b, nWqkv);
    cvt(Wo,   wo_b,   nWo);
    cvt(Wff1, wff1_b, nWff1);
    cvt(Wff2, wff2_b, nWff2);
    cvt(Wcls, wcls_b, nWcls);

    embed_ln_kernel<<<BT, 256, 0, stream>>>(input_ids, tti, we, pe, te, egs, egb, x_f, x_b);

    auto gemm = [&](const bf16_t* A, const bf16_t* B, void* C, const float* bias,
                    int M, int N, int K, int lda, int ldb, int ldc,
                    int nz, int zInner, long aso, long asi, long bso, long bsi,
                    long cso, long csi, int flags) {
        dim3 g((N + 127) / 128, (M + 127) / 128, nz);
        gemm_bf16_wmma<<<g, 256, 0, stream>>>(A, B, C, bias, M, N, K, lda, ldb, ldc,
                                              zInner, aso, asi, bso, bsi, cso, csi, flags);
    };

    for (int l = 0; l < LL; l++) {
        // QKV projection: [8192,768] @ [768,2304] -> bf16 qkv
        gemm(x_b, wqkv_b + (long)l * HH * 3 * HH, qkv_b, bqkv + (long)l * 3 * HH,
             BT, 3 * HH, HH, HH, 3 * HH, 3 * HH,
             1, 1, 0, 0, 0, 0, 0, 0, GF_BIAS | GF_BF16);

        // scores = Q @ K^T per (b,h): M=512,N=512,K=64  (z = b*12 + h)
        gemm(qkv_b, qkv_b + HH, scores, nullptr,
             TT, TT, DHH, 3 * HH, 3 * HH, TT,
             BB * NHH, NHH,
             (long)TT * 3 * HH, DHH,                 // A: +b*T*2304 + h*64
             (long)TT * 3 * HH, DHH,                 // B: +b*T*2304 + h*64 (K base)
             (long)NHH * TT * TT, (long)TT * TT,     // C: +(b*12+h)*T*T
             GF_BT);

        softmax_kernel<<<BB * NHH * TT, 256, 0, stream>>>(scores, p_b, attn, 0.125f);

        // ctx = P @ V per (b,h): M=512,N=64,K=512 -> ctx[b,t,h*64+dh] bf16
        gemm(p_b, qkv_b + 2 * HH, ctx_b, nullptr,
             TT, DHH, TT, TT, 3 * HH, HH,
             BB * NHH, NHH,
             (long)NHH * TT * TT, (long)TT * TT,
             (long)TT * 3 * HH, DHH,
             (long)TT * HH, DHH,
             GF_BF16);

        // attention output projection
        gemm(ctx_b, wo_b + (long)l * HH * HH, y_f, bo + (long)l * HH,
             BT, HH, HH, HH, HH, HH, 1, 1, 0, 0, 0, 0, 0, 0, GF_BIAS);
        residual_ln_kernel<<<BT, 256, 0, stream>>>(y_f, l1s + (long)l * HH, l1b + (long)l * HH, x_f, x_b);

        // FFN
        gemm(x_b, wff1_b + (long)l * HH * DFFF, h_b, bff1 + (long)l * DFFF,
             BT, DFFF, HH, HH, DFFF, DFFF, 1, 1, 0, 0, 0, 0, 0, 0,
             GF_BIAS | GF_GELU | GF_BF16);
        gemm(h_b, wff2_b + (long)l * DFFF * HH, y_f, bff2 + (long)l * HH,
             BT, HH, DFFF, DFFF, HH, HH, 1, 1, 0, 0, 0, 0, 0, 0, GF_BIAS);
        residual_ln_kernel<<<BT, 256, 0, stream>>>(y_f, l2s + (long)l * HH, l2b + (long)l * HH, x_f, x_b);
    }

    // classifier: [8192,768] @ [768,9] + bcls
    gemm(x_b, wcls_b, logits, bcls, BT, NLL, HH, HH, NLL, NLL,
         1, 1, 0, 0, 0, 0, 0, 0, GF_BIAS);

    crf_kernel<<<1, 32, 0, stream>>>(logits, labels, attn, cst, cen, ctr,
                                     (float*)d_out, hist);
}
